// MLP_LSTM_49778670960726
// MI455X (gfx1250) — matrix operand order repaired and checked by reading, not compiled, not run on hardware
//
#include <hip/hip_runtime.h>
#include <hip/hip_bf16.h>

// Problem constants (match reference)
#define BB   128
#define SS   512
#define DIN  512
#define DH   512
#define GG   (4 * DH)
#define BSZ  (BB * SS)          // 65536 rows for the big GEMMs

typedef __bf16 bf16;
typedef __attribute__((ext_vector_type(16))) bf16  v16bf;
typedef __attribute__((ext_vector_type(8)))  bf16  v8bf;
typedef __attribute__((ext_vector_type(8)))  float v8f;

// ---------------------------------------------------------------------------
// WMMA helpers (layouts per CDNA5 ISA §7.12.2, wave32)
// ---------------------------------------------------------------------------

// A fragment: 16x32 bf16, row M = lane%16.
// lanes 0-15 : elems 0..7 = K(k0..k0+7),   elems 8..15 = K(k0+16..k0+23)
// lanes16-31 : elems 0..7 = K(k0+8..k0+15),elems 8..15 = K(k0+24..k0+31)
static __device__ __forceinline__ v16bf load_a_frag(const bf16* __restrict__ A,
                                                    int lda, int row0, int k0, int lane) {
    const int m  = row0 + (lane & 15);
    const int kh = (lane >> 4) << 3;                 // 0 or 8
    const bf16* p = A + (size_t)m * lda + k0 + kh;
    v8bf lo = *reinterpret_cast<const v8bf*>(p);       // 16B load
    v8bf hi = *reinterpret_cast<const v8bf*>(p + 16);  // 16B load
    v16bf r;
#pragma unroll
    for (int i = 0; i < 8; ++i) { r[i] = lo[i]; r[8 + i] = hi[i]; }
    return r;
}

// B fragment: 32x16 bf16 with B[k][n] = W[n][k] (W stored [N x K] row-major).
// lane holds column n = lane%16; lanes 0-15 carry K(k0..k0+7, k0+16..k0+23),
// lanes 16-31 carry K(k0+8..k0+15, k0+24..k0+31) (same order as A).
static __device__ __forceinline__ v16bf load_b_frag(const bf16* __restrict__ W,
                                                    int ldw, int col0, int k0, int lane) {
    const int n  = col0 + (lane & 15);
    const int ko = (lane >> 4) << 3;                 // 0 or 8
    const bf16* p = W + (size_t)n * ldw + k0 + ko;
    v8bf lo = *reinterpret_cast<const v8bf*>(p);
    v8bf hi = *reinterpret_cast<const v8bf*>(p + 16);
    v16bf r;
#pragma unroll
    for (int i = 0; i < 8; ++i) { r[i] = lo[i]; r[8 + i] = hi[i]; }
    return r;
}

static __device__ __forceinline__ v8f wmma_bf16(v16bf a, v16bf b, v8f c) {
    return __builtin_amdgcn_wmma_f32_16x16x32_bf16(
        /*neg_a=*/false, a, /*neg_b=*/false, b,
        /*c_mod=*/(short)0, c, /*reuse_a=*/false, /*reuse_b=*/false);
}

static __device__ __forceinline__ float sigmoid_f(float x) {
    return 1.0f / (1.0f + __expf(-x));
}

// ---------------------------------------------------------------------------
// Elementwise prep kernels
// ---------------------------------------------------------------------------
__global__ void cvt_f32_bf16(const float* __restrict__ in, bf16* __restrict__ out, int n) {
    int i = blockIdx.x * blockDim.x + threadIdx.x;
    int stride = gridDim.x * blockDim.x;
    for (; i < n; i += stride) out[i] = (bf16)in[i];
}

__global__ void bias_sum_k(const float* __restrict__ a, const float* __restrict__ b,
                           float* __restrict__ o, int n) {
    int i = blockIdx.x * blockDim.x + threadIdx.x;
    if (i < n) o[i] = a[i] + b[i];
}

__global__ void copy_f32(const float* __restrict__ in, float* __restrict__ out, int n) {
    int i = blockIdx.x * blockDim.x + threadIdx.x;
    if (i < n) out[i] = in[i];
}

// ---------------------------------------------------------------------------
// Phase 1: u = relu(x @ W_l^T + b_l)   [BSZ x DIN] x [DH x DIN]^T -> bf16 U
// One 16x16 tile per wave; 8 waves per 256-thread block.
// ---------------------------------------------------------------------------
__global__ void mlp_gemm(const bf16* __restrict__ X, const bf16* __restrict__ Wl,
                         const float* __restrict__ bl, bf16* __restrict__ U) {
    const int lane = threadIdx.x & 31;
    const int wid  = threadIdx.x >> 5;
    const int NT   = DH / 16;                         // 32 column tiles
    const int tile = blockIdx.x * 8 + wid;
    const int mt = tile / NT, nt = tile % NT;
    const int row0 = mt * 16, col0 = nt * 16;

    v8f acc = {};
#pragma unroll 4
    for (int k = 0; k < DIN; k += 32) {
        v16bf a = load_a_frag(X,  DIN, row0, k, lane);
        v16bf b = load_b_frag(Wl, DIN, col0, k, lane);
        acc = wmma_bf16(a, b, acc);
    }
    const int n     = col0 + (lane & 15);
    const float bia = bl[n];
    const int mbase = row0 + ((lane >> 4) << 3);      // C layout: M = r + 8*(lane/16)
#pragma unroll
    for (int r = 0; r < 8; ++r) {
        float v = acc[r] + bia;
        v = v > 0.0f ? v : 0.0f;
        U[(size_t)(mbase + r) * DH + n] = (bf16)v;
    }
}

// ---------------------------------------------------------------------------
// Phase 2: gx = u @ W_ih^T + (b_ih + b_hh), stored transposed as [S][B][4H] bf16
// ---------------------------------------------------------------------------
__global__ void gx_gemm(const bf16* __restrict__ U, const bf16* __restrict__ Wih,
                        const float* __restrict__ bsum, bf16* __restrict__ GX) {
    const int lane = threadIdx.x & 31;
    const int wid  = threadIdx.x >> 5;
    const int NT   = GG / 16;                         // 128 column tiles
    const int tile = blockIdx.x * 8 + wid;
    const int mt = tile / NT, nt = tile % NT;
    const int row0 = mt * 16, col0 = nt * 16;

    v8f acc = {};
#pragma unroll 4
    for (int k = 0; k < DH; k += 32) {
        v16bf a = load_a_frag(U,   DH, row0, k, lane);
        v16bf b = load_b_frag(Wih, DH, col0, k, lane);
        acc = wmma_bf16(a, b, acc);
    }
    const int n     = col0 + (lane & 15);
    const float bia = bsum[n];
    const int mbase = row0 + ((lane >> 4) << 3);
#pragma unroll
    for (int r = 0; r < 8; ++r) {
        const int m    = mbase + r;                   // m = b*S + s
        const int bidx = m / SS, sidx = m % SS;
        GX[((size_t)sidx * BB + bidx) * GG + n] = (bf16)(acc[r] + bia);
    }
}

// ---------------------------------------------------------------------------
// Phase 3: one LSTM step. gates = gx[t] + h @ W_hh^T ; h,c update.
// 8 M-tiles x 32 N-tiles = 256 wave-tiles; each wave owns one 16x16 h-column
// tile and accumulates all 4 gates (cols n, n+512, n+1024, n+1536) so the
// pointwise update is local. Grid sync = kernel boundary (one launch / step).
// ---------------------------------------------------------------------------
__global__ void lstm_step(const bf16* __restrict__ GXt, const bf16* __restrict__ Whh,
                          const bf16* __restrict__ Hin, bf16* __restrict__ Hout,
                          float* __restrict__ Hf, float* __restrict__ C) {
    const int lane = threadIdx.x & 31;
    const int wid  = threadIdx.x >> 5;
    const int tile = blockIdx.x * 8 + wid;            // 0..255
    const int mt = tile >> 5, nt = tile & 31;
    const int row0 = mt * 16, col0 = nt * 16;

    v8f ai = {}, af = {}, ag = {}, ao = {};
#pragma unroll 2
    for (int k = 0; k < DH; k += 32) {
        v16bf a  = load_a_frag(Hin, DH, row0, k, lane);
        v16bf bi = load_b_frag(Whh, DH, col0 + 0 * DH, k, lane);
        v16bf bf_ = load_b_frag(Whh, DH, col0 + 1 * DH, k, lane);
        v16bf bg = load_b_frag(Whh, DH, col0 + 2 * DH, k, lane);
        v16bf bo = load_b_frag(Whh, DH, col0 + 3 * DH, k, lane);
        ai = wmma_bf16(a, bi,  ai);
        af = wmma_bf16(a, bf_, af);
        ag = wmma_bf16(a, bg,  ag);
        ao = wmma_bf16(a, bo,  ao);
    }

    const int n     = col0 + (lane & 15);
    const int mbase = row0 + ((lane >> 4) << 3);
#pragma unroll
    for (int r = 0; r < 8; ++r) {
        const int b = mbase + r;
        const size_t gbase = (size_t)b * GG + n;
        float iv = ai[r] + (float)GXt[gbase + 0 * DH];
        float fv = af[r] + (float)GXt[gbase + 1 * DH];
        float gv = ag[r] + (float)GXt[gbase + 2 * DH];
        float ov = ao[r] + (float)GXt[gbase + 3 * DH];
        iv = sigmoid_f(iv);
        fv = sigmoid_f(fv);
        gv = tanhf(gv);
        ov = sigmoid_f(ov);
        const size_t hidx = (size_t)b * DH + n;
        const float cn = fv * C[hidx] + iv * gv;
        const float hn = ov * tanhf(cn);
        C[hidx]    = cn;
        Hout[hidx] = (bf16)hn;                        // next step's A matrix
        Hf[hidx]   = hn;                              // f32 copy for output
    }
}

// ---------------------------------------------------------------------------
// Assemble d_out = [h, h, c] (reference returns h, (h[None], c[None]))
// ---------------------------------------------------------------------------
__global__ void finalize(const float* __restrict__ H, const float* __restrict__ C,
                         float* __restrict__ out) {
    int i = blockIdx.x * blockDim.x + threadIdx.x;
    if (i < BB * DH) {
        out[i]               = H[i];
        out[BB * DH + i]     = H[i];
        out[2 * BB * DH + i] = C[i];
    }
}

// ---------------------------------------------------------------------------
extern "C" void kernel_launch(void* const* d_in, const int* in_sizes, int n_in,
                              void* d_out, int out_size, void* d_ws, size_t ws_size,
                              hipStream_t stream) {
    (void)in_sizes; (void)n_in; (void)out_size; (void)ws_size;
    const float* x_in = (const float*)d_in[0];
    const float* W_l  = (const float*)d_in[1];
    const float* b_l  = (const float*)d_in[2];
    const float* W_ih = (const float*)d_in[3];
    const float* W_hh = (const float*)d_in[4];
    const float* b_ih = (const float*)d_in[5];
    const float* b_hh = (const float*)d_in[6];
    const float* h0   = (const float*)d_in[7];
    const float* c0   = (const float*)d_in[8];

    char* ws = (char*)d_ws;
    size_t off = 0;
    auto alloc = [&](size_t bytes) -> char* {
        char* p = ws + off;
        off += (bytes + 255) & ~(size_t)255;
        return p;
    };

    bf16*  Xb    = (bf16*)alloc((size_t)BSZ * DIN * sizeof(bf16));   //  64 MB
    bf16*  Wlb   = (bf16*)alloc((size_t)DH * DIN * sizeof(bf16));    // 512 KB
    bf16*  Wihb  = (bf16*)alloc((size_t)GG * DH * sizeof(bf16));     //   2 MB
    bf16*  Whhb  = (bf16*)alloc((size_t)GG * DH * sizeof(bf16));     //   2 MB
    float* Bsum  = (float*)alloc((size_t)GG * sizeof(float));
    bf16*  Ub    = (bf16*)alloc((size_t)BSZ * DH * sizeof(bf16));    //  64 MB
    bf16*  GXb   = (bf16*)alloc((size_t)BSZ * GG * sizeof(bf16));    // 256 MB
    bf16*  Hb0   = (bf16*)alloc((size_t)BB * DH * sizeof(bf16));
    bf16*  Hb1   = (bf16*)alloc((size_t)BB * DH * sizeof(bf16));
    float* Hf    = (float*)alloc((size_t)BB * DH * sizeof(float));
    float* Cf    = (float*)alloc((size_t)BB * DH * sizeof(float));

    // --- prep: bf16 conversions, bias sum, state init ---
    cvt_f32_bf16<<<4096, 256, 0, stream>>>(x_in, Xb,  BSZ * DIN);
    cvt_f32_bf16<<<1024, 256, 0, stream>>>(W_l,  Wlb, DH * DIN);
    cvt_f32_bf16<<<4096, 256, 0, stream>>>(W_ih, Wihb, GG * DH);
    cvt_f32_bf16<<<4096, 256, 0, stream>>>(W_hh, Whhb, GG * DH);
    bias_sum_k<<<(GG + 255) / 256, 256, 0, stream>>>(b_ih, b_hh, Bsum, GG);
    cvt_f32_bf16<<<256, 256, 0, stream>>>(h0, Hb0, BB * DH);
    copy_f32<<<256, 256, 0, stream>>>(h0, Hf, BB * DH);
    copy_f32<<<256, 256, 0, stream>>>(c0, Cf, BB * DH);

    // --- phase 1: MLP GEMM ---  (65536/16)*(512/16) = 131072 tiles, 8/block
    mlp_gemm<<<16384, 256, 0, stream>>>(Xb, Wlb, b_l, Ub);

    // --- phase 2: input-gate GEMM ---  (65536/16)*(2048/16) = 524288 tiles
    gx_gemm<<<65536, 256, 0, stream>>>(Ub, Wihb, Bsum, GXb);

    // --- phase 3: recurrence, one kernel per step (kernel boundary = grid sync)
    for (int t = 0; t < SS; ++t) {
        const bf16* GXt = GXb + (size_t)t * BB * GG;
        bf16* hin  = (t & 1) ? Hb1 : Hb0;
        bf16* hout = (t & 1) ? Hb0 : Hb1;
        lstm_step<<<32, 256, 0, stream>>>(GXt, Whhb, hin, hout, Hf, Cf);
    }

    // --- output: [h, h, c] ---
    finalize<<<(BB * DH + 255) / 256, 256, 0, stream>>>(Hf, Cf, (float*)d_out);
}
